// ContentGuidedAttention_19464791786092
// MI455X (gfx1250) — compile-verified, matrix operand order checked by reading, not run on hardware
//
#include <hip/hip_runtime.h>
#include <hip/hip_bf16.h>

// ---------------------------------------------------------------------------
// ContentGuidedAttention for MI455X (gfx1250).
// B=4, C=256, N=4096.  ~77 GFLOP over ~70MB HBM traffic -> compute bound:
// all GEMMs on v_wmma_f32_16x16x32_bf16 (f32 accumulate), operands pre-
// converted to bf16, QKV staged through LDS via async global->LDS copies.
// ---------------------------------------------------------------------------

#define B_  4
#define C_  256
#define N_  4096
#define SCALE_ 0.17677669529663687f   // (256/8)^-0.5
#define LN_EPS_ 1e-5f

typedef __attribute__((ext_vector_type(8)))  float          v8f;
typedef __attribute__((ext_vector_type(8)))  unsigned short v8us;
typedef __attribute__((ext_vector_type(16))) unsigned short v16us;
typedef __attribute__((ext_vector_type(16))) __bf16         v16bf;

static __device__ inline unsigned short f2b(float f) {
    return __builtin_bit_cast(unsigned short, (__bf16)f);   // RNE fptrunc
}
static __device__ inline float b2f(unsigned short h) {
    unsigned u = (unsigned)h << 16;
    return __builtin_bit_cast(float, u);
}
static __device__ inline v16bf cat8(v8us lo, v8us hi) {
    v16us w = __builtin_shufflevector(lo, hi, 0,1,2,3,4,5,6,7,8,9,10,11,12,13,14,15);
    return __builtin_bit_cast(v16bf, w);
}
static __device__ inline v8f wmma_bf16(v16bf a, v16bf b, v8f c) {
    return __builtin_amdgcn_wmma_f32_16x16x32_bf16(
        false, a, false, b, (short)0, c, false, false);
}

// --- gfx1250 async global->LDS path (ASYNCcnt), with portable fallback -----
#if defined(__HIP_DEVICE_COMPILE__) && \
    __has_builtin(__builtin_amdgcn_global_load_async_to_lds_b128) && \
    __has_builtin(__builtin_amdgcn_s_wait_asynccnt)
#define CGA_ASYNC_LDS 1
typedef int cga_v4i __attribute__((vector_size(4 * sizeof(int))));
typedef __attribute__((address_space(3))) cga_v4i* cga_l128p;
#endif

static __device__ inline void g2l_16B(const unsigned short* g, unsigned short* l) {
#ifdef CGA_ASYNC_LDS
    __builtin_amdgcn_global_load_async_to_lds_b128(
        (cga_v4i*)g, (cga_l128p)l, 0, 0);
#else
    *(v8us*)l = *(const v8us*)g;
#endif
}
static __device__ inline void cga_wait_async() {
#ifdef CGA_ASYNC_LDS
    __builtin_amdgcn_s_wait_asynccnt(0);
#endif
}

// ---------------------------------------------------------------------------
// Kernel 0: one-shot f32 -> bf16 conversion of activations and weights.
// grid = (2048, 6): y selects {low, high, q_w, k_w, v_w, o_w}; 8 elems/thread.
// ---------------------------------------------------------------------------
__global__ __launch_bounds__(256) void cga_cvt_kernel(
    const float* __restrict__ low, const float* __restrict__ high,
    const float* __restrict__ qw,  const float* __restrict__ kw,
    const float* __restrict__ vw,  const float* __restrict__ ow,
    unsigned short* __restrict__ lowb, unsigned short* __restrict__ highb,
    unsigned short* __restrict__ qwb,  unsigned short* __restrict__ kwb,
    unsigned short* __restrict__ vwb,  unsigned short* __restrict__ owb)
{
    const size_t big = (size_t)B_ * C_ * N_;   // 4,194,304
    const size_t wsz = (size_t)C_ * C_;        // 65,536
    const float* src; unsigned short* dst; size_t count;
    switch (blockIdx.y) {
        case 0:  src = low;  dst = lowb;  count = big; break;
        case 1:  src = high; dst = highb; count = big; break;
        case 2:  src = qw;   dst = qwb;   count = wsz; break;
        case 3:  src = kw;   dst = kwb;   count = wsz; break;
        case 4:  src = vw;   dst = vwb;   count = wsz; break;
        default: src = ow;   dst = owb;   count = wsz; break;
    }
    const size_t base = ((size_t)blockIdx.x * 256 + threadIdx.x) * 8;
    if (base >= count) return;
    const float4 a = ((const float4*)(src + base))[0];
    const float4 b = ((const float4*)(src + base))[1];
    v8us o;
    o[0] = f2b(a.x); o[1] = f2b(a.y); o[2] = f2b(a.z); o[3] = f2b(a.w);
    o[4] = f2b(b.x); o[5] = f2b(b.y); o[6] = f2b(b.z); o[7] = f2b(b.w);
    *(v8us*)(dst + base) = o;
}

// ---------------------------------------------------------------------------
// Kernel 1: QKV projection, workgroup-cooperative.
// Block = 256 threads computes 16(out-ch) x 128(tokens): wave w owns the
// 16x16 tile at tokens nblk+16w.  Per K-step the 32x128 bf16 activation slab
// is staged into LDS with coalesced 16B async global->LDS copies, then lanes
// assemble B fragments from LDS (conflict-free consecutive-lane reads).
// Q,K written token-major bf16 (B,N,C); V channel-major bf16 (B,C,N).
// grid = (N/128, C/16, 3*B), block = 256.
// ---------------------------------------------------------------------------
__global__ __launch_bounds__(256) void cga_qkv_kernel(
    const unsigned short* __restrict__ lowb, const unsigned short* __restrict__ highb,
    const unsigned short* __restrict__ qwb,  const unsigned short* __restrict__ kwb,
    const unsigned short* __restrict__ vwb,
    const float* __restrict__ q_b, const float* __restrict__ k_b,
    const float* __restrict__ v_b,
    unsigned short* __restrict__ Qo, unsigned short* __restrict__ Ko,
    unsigned short* __restrict__ Vo)
{
    __shared__ unsigned short tile[32 * 128];    // 8KB K-slab x tokens
    const int tid  = threadIdx.x;
    const int lane = tid & 31;
    const int wave = tid >> 5;
    const int ml   = lane & 15;
    const int half = lane >> 4;
    const int nblk = blockIdx.x * 128;
    const int c0   = blockIdx.y * 16;
    const int z    = blockIdx.z;
    const int which = z / B_;            // 0=Q(low) 1=K(high) 2=V(high)
    const int b     = z % B_;

    const unsigned short* Wb; const float* bias; const unsigned short* Xb;
    if (which == 0)      { Wb = qwb; bias = q_b; Xb = lowb;  }
    else if (which == 1) { Wb = kwb; bias = k_b; Xb = highb; }
    else                 { Wb = vwb; bias = v_b; Xb = highb; }
    Xb += (size_t)b * C_ * N_;

    const int srow = tid >> 3;           // 0..31 : slab row this thread copies
    const int scol = (tid & 7) * 16;     // 0,16,...,112
    const int cl   = wave * 16 + ml;     // this lane's token column in slab

    v8f acc = {};
    const unsigned short* arow = Wb + (size_t)(c0 + ml) * C_;
    for (int kk = 0; kk < C_; kk += 32) {
        __syncthreads();                 // previous slab fully consumed
        const unsigned short* src = Xb + (size_t)(kk + srow) * N_ + nblk + scol;
        unsigned short*       dst = tile + srow * 128 + scol;
        g2l_16B(src,     dst);
        g2l_16B(src + 8, dst + 8);
        if (kk + 32 < C_) __builtin_prefetch(src + 32 * N_, 0, 0);
        cga_wait_async();
        __syncthreads();                 // slab visible to all waves

        const int kb = half * 8;         // per-lane K sub-offset in slab
        v16us bu;
        #pragma unroll
        for (int e = 0; e < 8; ++e) {
            bu[e]     = tile[(kb + e)      * 128 + cl];
            bu[e + 8] = tile[(kb + 16 + e) * 128 + cl];
        }
        const int ka = kk + kb;
        acc = wmma_bf16(cat8(*(const v8us*)(arow + ka),
                             *(const v8us*)(arow + ka + 16)),
                        __builtin_bit_cast(v16bf, bu), acc);
    }
    #pragma unroll
    for (int r = 0; r < 8; ++r) {
        const int ch  = c0 + r + half * 8;       // D: VGPR r -> M=r (+8 upper)
        const int tok = nblk + wave * 16 + ml;   //    N = lane%16
        const unsigned short bv = f2b(acc[r] + bias[ch]);
        if (which == 2)
            Vo[((size_t)b * C_ + ch) * N_ + tok] = bv;   // V^T (C,N) contiguous
        else {
            unsigned short* D = (which == 0) ? Qo : Ko;
            D[((size_t)b * N_ + tok) * C_ + ch] = bv;    // token-major
        }
    }
}

// ---------------------------------------------------------------------------
// Kernel 2: attention.  One workgroup (8 waves) per 16-query tile.
// Scores in LDS as bf16: 16 x 4096 x 2B = 128KB of the 320KB/WGP.
// grid = B*N/16 = 1024 blocks, block = 256, dynLDS = 128KB.
// ---------------------------------------------------------------------------
__global__ __launch_bounds__(256) void cga_attn_kernel(
    const unsigned short* __restrict__ Q,
    const unsigned short* __restrict__ K,
    const unsigned short* __restrict__ V,
    unsigned short* __restrict__ O)
{
    extern __shared__ unsigned short S[];        // 16 * 4096 bf16
    const int tid  = threadIdx.x;
    const int lane = tid & 31;
    const int wave = tid >> 5;
    const int ml   = lane & 15;
    const int half = lane >> 4;
    const int b    = blockIdx.x >> 8;
    const int q0   = (blockIdx.x & 255) * 16;

    // Q fragments for the whole C loop (A operand, resident in VGPRs)
    v16bf aq[8];
    const unsigned short* qrow = Q + ((size_t)b * N_ + q0 + ml) * C_;
    #pragma unroll
    for (int ks = 0; ks < 8; ++ks) {
        const int kb = ks * 32 + half * 8;
        aq[ks] = cat8(*(const v8us*)(qrow + kb), *(const v8us*)(qrow + kb + 16));
    }

    // ---- Step 1: S = scale * Q K^T  (keys striped across waves) --------
    const unsigned short* Kb = K + (size_t)b * N_ * C_;
    for (int mt = wave; mt < N_ / 16; mt += 8) {
        const int m0 = mt * 16;
        const unsigned short* krow = Kb + (size_t)(m0 + ml) * C_;
        __builtin_prefetch(krow + 128 * C_, 0, 0);
        v8f acc = {};
        #pragma unroll
        for (int ks = 0; ks < 8; ++ks) {
            const int kb = ks * 32 + half * 8;
            acc = wmma_bf16(aq[ks],
                            cat8(*(const v8us*)(krow + kb),
                                 *(const v8us*)(krow + kb + 16)), acc);
        }
        #pragma unroll
        for (int r = 0; r < 8; ++r)
            S[(size_t)(r + half * 8) * N_ + m0 + ml] = f2b(acc[r] * SCALE_);
    }
    __syncthreads();

    // ---- Step 2: softmax, 16 lanes per row ------------------------------
    {
        const int r = tid >> 4, i = tid & 15;
        unsigned short* row = S + (size_t)r * N_;
        float mx = -3.0e38f;
        for (int j = i; j < N_; j += 16) mx = fmaxf(mx, b2f(row[j]));
        #pragma unroll
        for (int m = 8; m >= 1; m >>= 1) mx = fmaxf(mx, __shfl_xor(mx, m, 32));
        float sum = 0.f;
        for (int j = i; j < N_; j += 16) {
            const float e = __expf(b2f(row[j]) - mx);
            row[j] = f2b(e); sum += e;
        }
        #pragma unroll
        for (int m = 8; m >= 1; m >>= 1) sum += __shfl_xor(sum, m, 32);
        const float inv = 1.0f / sum;
        for (int j = i; j < N_; j += 16) row[j] = f2b(b2f(row[j]) * inv);
    }
    __syncthreads();

    // ---- Step 3: O = P V  (wave owns 32 output channels); A from LDS ----
    const unsigned short* Vb = V + (size_t)b * C_ * N_;
    const int cb = wave * 32;
    v8f acc0 = {}, acc1 = {};
    const unsigned short* srow = S + (size_t)ml * N_;
    for (int kk = 0; kk < N_; kk += 32) {
        const int kb = kk + half * 8;
        const v16bf a = cat8(*(const v8us*)(srow + kb),
                             *(const v8us*)(srow + kb + 16));   // ds_load_b128
        const unsigned short* v0 = Vb + (size_t)(cb + ml)      * N_ + kb;
        const unsigned short* v1 = Vb + (size_t)(cb + 16 + ml) * N_ + kb;
        __builtin_prefetch(v0 + 256, 0, 0);
        acc0 = wmma_bf16(a, cat8(*(const v8us*)v0, *(const v8us*)(v0 + 16)), acc0);
        acc1 = wmma_bf16(a, cat8(*(const v8us*)v1, *(const v8us*)(v1 + 16)), acc1);
    }
    unsigned short* Ob = O + (size_t)b * N_ * C_;
    #pragma unroll
    for (int r = 0; r < 8; ++r) {
        const int tok = q0 + r + half * 8;
        Ob[(size_t)tok * C_ + cb + ml]      = f2b(acc0[r]);
        Ob[(size_t)tok * C_ + cb + 16 + ml] = f2b(acc1[r]);
    }
}

// ---------------------------------------------------------------------------
// Kernel 3: out-projection + residual + LayerNorm.  One wave owns 16 tokens
// x 256 channels (16 accumulator tiles) so LN reduces via 16-lane shfl_xor.
// grid = 128 blocks, block = 256.
// ---------------------------------------------------------------------------
__global__ __launch_bounds__(256) void cga_oln_kernel(
    const unsigned short* __restrict__ O,   const unsigned short* __restrict__ owb,
    const float* __restrict__ low,  const float* __restrict__ o_b,
    const float* __restrict__ ln_g, const float* __restrict__ ln_b,
    float* __restrict__ out)
{
    const int tid  = threadIdx.x;
    const int lane = tid & 31;
    const int wave = tid >> 5;
    const int tile = blockIdx.x * 8 + wave;     // 0..1023
    const int b    = tile >> 8;
    const int n0   = (tile & 255) * 16;
    const int ml   = lane & 15;
    const int half = lane >> 4;

    v8f acc[16];
    #pragma unroll
    for (int ct = 0; ct < 16; ++ct) acc[ct] = (v8f){};

    const unsigned short* Orow = O + ((size_t)b * N_ + n0 + ml) * C_;
    for (int ks = 0; ks < 8; ++ks) {
        const int kb = ks * 32 + half * 8;
        const v16bf a = cat8(*(const v8us*)(Orow + kb),
                             *(const v8us*)(Orow + kb + 16));
        #pragma unroll
        for (int ct = 0; ct < 16; ++ct) {
            // B: o_w^T -> lane holds out-ch ct*16+ml, K runs over in-ch
            const unsigned short* wrow = owb + (size_t)(ct * 16 + ml) * C_;
            acc[ct] = wmma_bf16(a, cat8(*(const v8us*)(wrow + kb),
                                        *(const v8us*)(wrow + kb + 16)), acc[ct]);
        }
    }

    // bias + residual, then LN over channels
    const float* lowbp = low + (size_t)b * C_ * N_;
    #pragma unroll
    for (int ct = 0; ct < 16; ++ct) {
        const int ch = ct * 16 + ml;
        const float bias = o_b[ch];
        #pragma unroll
        for (int r = 0; r < 8; ++r) {
            const int tok = n0 + r + half * 8;
            acc[ct][r] += bias + lowbp[(size_t)ch * N_ + tok];
        }
    }
    #pragma unroll
    for (int r = 0; r < 8; ++r) {
        float s = 0.f, ss = 0.f;
        #pragma unroll
        for (int ct = 0; ct < 16; ++ct) { const float y = acc[ct][r]; s += y; ss += y * y; }
        #pragma unroll
        for (int m = 8; m >= 1; m >>= 1) {
            s  += __shfl_xor(s,  m, 32);
            ss += __shfl_xor(ss, m, 32);
        }
        const float mu  = s * (1.0f / 256.0f);
        const float var = ss * (1.0f / 256.0f) - mu * mu;
        const float inv = rsqrtf(var + LN_EPS_);
        const int tok = n0 + r + half * 8;
        #pragma unroll
        for (int ct = 0; ct < 16; ++ct) {
            const int ch = ct * 16 + ml;
            out[((size_t)b * C_ + ch) * N_ + tok] =
                (acc[ct][r] - mu) * inv * ln_g[ch] + ln_b[ch];
        }
    }
}

// ---------------------------------------------------------------------------
extern "C" void kernel_launch(void* const* d_in, const int* in_sizes, int n_in,
                              void* d_out, int out_size, void* d_ws, size_t ws_size,
                              hipStream_t stream) {
    (void)in_sizes; (void)n_in; (void)out_size; (void)ws_size;
    const float* low  = (const float*)d_in[0];
    const float* high = (const float*)d_in[1];
    const float* q_w  = (const float*)d_in[2];
    const float* q_b  = (const float*)d_in[3];
    const float* k_w  = (const float*)d_in[4];
    const float* k_b  = (const float*)d_in[5];
    const float* v_w  = (const float*)d_in[6];
    const float* v_b  = (const float*)d_in[7];
    const float* o_w  = (const float*)d_in[8];
    const float* o_b  = (const float*)d_in[9];
    const float* ln_g = (const float*)d_in[10];
    const float* ln_b = (const float*)d_in[11];

    // workspace (bf16): lowb | highb | qwb | kwb | vwb | owb | Q | K | Vt | O
    const size_t big = (size_t)B_ * C_ * N_;   // 4,194,304 elements
    const size_t wsz = (size_t)C_ * C_;        // 65,536 elements
    unsigned short* lowb  = (unsigned short*)d_ws;
    unsigned short* highb = lowb  + big;
    unsigned short* qwb   = highb + big;
    unsigned short* kwb   = qwb   + wsz;
    unsigned short* vwb   = kwb   + wsz;
    unsigned short* owb   = vwb   + wsz;
    unsigned short* Q     = owb   + wsz;
    unsigned short* Kt    = Q     + big;
    unsigned short* Vt    = Kt    + big;
    unsigned short* Obf   = Vt    + big;

    cga_cvt_kernel<<<dim3(2048, 6), 256, 0, stream>>>(
        low, high, q_w, k_w, v_w, o_w, lowb, highb, qwb, kwb, vwb, owb);

    cga_qkv_kernel<<<dim3(N_ / 128, C_ / 16, 3 * B_), 256, 0, stream>>>(
        lowb, highb, qwb, kwb, vwb, q_b, k_b, v_b, Q, Kt, Vt);

    const size_t lds = (size_t)16 * N_ * sizeof(unsigned short);  // 128KB
    (void)hipFuncSetAttribute((const void*)cga_attn_kernel,
                              hipFuncAttributeMaxDynamicSharedMemorySize, (int)lds);
    cga_attn_kernel<<<dim3(B_ * (N_ / 16)), 256, lds, stream>>>(Q, Kt, Vt, Obf);

    cga_oln_kernel<<<dim3(B_ * (N_ / 16) / 8), 256, 0, stream>>>(
        Obf, owb, low, o_b, ln_g, ln_b, (float*)d_out);
}
